// GAT_25134148616267
// MI455X (gfx1250) — matrix operand order repaired and checked by reading, not compiled, not run on hardware
//
#include <hip/hip_runtime.h>
#include <hip/hip_bf16.h>

typedef __attribute__((ext_vector_type(16))) _Float16 v16h;
typedef __attribute__((ext_vector_type(8)))  _Float16 v8h;
typedef __attribute__((ext_vector_type(8)))  float    v8f;

#define NN      3072
#define NHEADSC 8
#define FOUTC   64
#define OUTW    (NHEADSC * FOUTC)   // 512
#define ALPHAC  0.2f
#define NEGF    (-9.0e15f)
#define NBLK    (NN / 32)           // 96 column blocks per head

__device__ __forceinline__ v8f wmma_f16(v16h a, v16h b, v8f c) {
  // D = A(16x32 f16) x B(32x16 f16) + C(16x16 f32)
  return __builtin_amdgcn_wmma_f32_16x16x32_f16(false, a, false, b, (short)0, c,
                                                false, false);
}

// Async global->LDS copy, 16B per lane, tracked by ASYNCcnt (gfx1250).
__device__ __forceinline__ void async_b128(uint32_t lds_addr, int gbyte_off,
                                           const void* sbase) {
  asm volatile("global_load_async_to_lds_b128 %0, %1, %2"
               :
               : "v"(lds_addr), "v"(gbyte_off), "s"(sbase)
               : "memory");
}

// ---------------------------------------------------------------------------
// Kernel 1: Wh = h @ W (per head) + e_src/e_dst row dots.
// One wave -> 16-row x 64-col Wh tile; 4 waves share the LDS-staged W tile.
// Wh stored pre-transposed in 32-row blocks: [head][blk][feat 0..63][col 0..31]
// so kernel 2's async copies land directly in B-operand layout.
// ---------------------------------------------------------------------------
__global__ __launch_bounds__(128)
void gat_wh_kernel(const float* __restrict__ h, const float* __restrict__ W,
                   const float* __restrict__ a, int fin,
                   _Float16* __restrict__ wh, float* __restrict__ esrc,
                   float* __restrict__ edst)
{
  __shared__ __align__(16) _Float16 wTT[64][40];  // [feature][k], padded stride

  const int wave = threadIdx.x >> 5;
  const int lane = threadIdx.x & 31;
  const int hi   = lane >> 4;
  const int lm   = lane & 15;
  const int hd   = blockIdx.y;
  const int m0   = blockIdx.x * 64 + wave * 16;

  v8f acc[4] = {};
  const float* hrow = h + (size_t)(m0 + lm) * fin;

  for (int kb = 0; kb < fin; kb += 32) {
    __syncthreads();
    // Stage W[hd][kb..kb+31][0..63] transposed into wTT[f][kk].
    const float* wsrc = W + ((size_t)hd * fin + kb) * FOUTC;
    #pragma unroll
    for (int i = 0; i < 16; ++i) {
      int idx = threadIdx.x * 16 + i;      // 0..2047
      int kk = idx >> 6, f = idx & 63;
      wTT[f][kk] = (_Float16)wsrc[kk * FOUTC + f];
    }
    __syncthreads();

    // A operand: row m0+lm; K = kb + 16*hi + j.
    const float4* p0 = (const float4*)(hrow + kb + 16 * hi);
    const float4* p1 = (const float4*)(hrow + kb + 16 * hi + 8);
    float4 r0 = p0[0], r1 = p0[1], r2 = p1[0], r3 = p1[1];
    float abuf[16] = {r0.x, r0.y, r0.z, r0.w, r1.x, r1.y, r1.z, r1.w,
                      r2.x, r2.y, r2.z, r2.w, r3.x, r3.y, r3.z, r3.w};
    v16h A;
    #pragma unroll
    for (int j = 0; j < 16; ++j) A[j] = (_Float16)abuf[j];

    // Hoist all B operands, then back-to-back WMMAs.
    v16h B[4];
    #pragma unroll
    for (int t = 0; t < 4; ++t) {
      int f = t * 16 + lm;
      v8h b0 = *(const v8h*)&wTT[f][16 * hi];
      v8h b1 = *(const v8h*)&wTT[f][16 * hi + 8];
      B[t] = __builtin_shufflevector(b0, b1, 0, 1, 2, 3, 4, 5, 6, 7,
                                     8, 9, 10, 11, 12, 13, 14, 15);
    }
    #pragma unroll
    for (int t = 0; t < 4; ++t) acc[t] = wmma_f16(A, B[t], acc[t]);
  }

  // Epilogue: store Wh (f16, blocked-transposed) + e_src/e_dst dots.
  const float* av = a + hd * (2 * FOUTC);
  float as_[4], ad_[4];
  #pragma unroll
  for (int t = 0; t < 4; ++t) {
    as_[t] = av[t * 16 + lm];
    ad_[t] = av[FOUTC + t * 16 + lm];
  }
  float ps[8], pd[8];
  #pragma unroll
  for (int r = 0; r < 8; ++r) {
    int row = m0 + 8 * hi + r;
    float s = 0.f, d = 0.f;
    #pragma unroll
    for (int t = 0; t < 4; ++t) {
      float v = acc[t][r];
      // [head][blk][feat][col]
      wh[((size_t)hd * NBLK + (row >> 5)) * 2048 + (t * 16 + lm) * 32 + (row & 31)]
          = (_Float16)v;
      s += v * as_[t];
      d += v * ad_[t];
    }
    ps[r] = s; pd[r] = d;
  }
  #pragma unroll
  for (int m = 1; m <= 8; m <<= 1) {
    #pragma unroll
    for (int r = 0; r < 8; ++r) {
      ps[r] += __shfl_xor(ps[r], m, 32);
      pd[r] += __shfl_xor(pd[r], m, 32);
    }
  }
  if (lm == 0) {
    #pragma unroll
    for (int r = 0; r < 8; ++r) {
      esrc[hd * NN + m0 + 8 * hi + r] = ps[r];
      edst[hd * NN + m0 + 8 * hi + r] = pd[r];
    }
  }
}

// ---------------------------------------------------------------------------
// Kernel 2: fused masked-softmax attention, flash style.
// 8 waves/block share one head; 64-column tiles double-buffered in LDS via
// async global->LDS copies (prefetch tile it+2 while computing tile it).
// ---------------------------------------------------------------------------
__global__ __launch_bounds__(256)
void gat_attn_kernel(const _Float16* __restrict__ whb,
                     const float* __restrict__ esrc,
                     const float* __restrict__ edst,
                     const float* __restrict__ adj,
                     float* __restrict__ out)
{
  // [buf][subtile][feat 0..63][col 0..31] f16 : 2 x 8KB
  __shared__ __align__(16) _Float16 whT[2][2][64][32];

  const int tid  = threadIdx.x;
  const int wave = tid >> 5;
  const int lane = tid & 31;
  const int hi   = lane >> 4;
  const int lm   = lane & 15;
  const int hd   = blockIdx.y;
  const int m0   = blockIdx.x * 128 + wave * 16;
  const int myrow = m0 + lm;

  // Per-thread LDS byte address for async copies (low 32 bits of generic ptr
  // to LDS == LDS byte offset per CDNA5 aperture rules).
  const uint32_t lds0 =
      (uint32_t)(uintptr_t)(&whT[0][0][0][0]) + (uint32_t)tid * 16u;
  const char* gwh = (const char*)(whb + (size_t)hd * NBLK * 2048);

  const float esr = esrc[hd * NN + myrow];
  const float* adjrow = adj + (size_t)myrow * NN;
  const float* edp = edst + hd * NN;

  float runmax = NEGF, runsum = 0.f;
  v8f acc[4] = {};

  const int NT = NN / 64;  // 48 tiles, 8KB each

  // Prologue: prefetch tiles 0 (buf0) and 1 (buf1): 2 async ops per tile/wave.
  async_b128(lds0,          tid * 16,          gwh);
  async_b128(lds0 + 4096u,  tid * 16 + 4096,   gwh);
  async_b128(lds0 + 8192u,  tid * 16 + 8192,   gwh);
  async_b128(lds0 + 12288u, tid * 16 + 12288,  gwh);

  for (int it = 0; it < NT; ++it) {
    const int buf = it & 1;
    // Current tile resident when <=2 newer ops remain in flight (in-order).
    if (it + 1 < NT) asm volatile("s_wait_asynccnt 0x2" ::: "memory");
    else             asm volatile("s_wait_asynccnt 0x0" ::: "memory");
    __syncthreads();

    const int cb = it * 64;

    // Logits for both 32-col subtiles in A-matrix lane layout:
    // lane owns row lm; element j -> col j+16*hi.
    float ev[2][16];
    #pragma unroll
    for (int s = 0; s < 2; ++s) {
      const float* ab = adjrow + cb + s * 32;
      const float* eb = edp    + cb + s * 32;
      float4 mq[4] = {((const float4*)(ab + 16 * hi))[0],
                      ((const float4*)(ab + 16 * hi))[1],
                      ((const float4*)(ab + 16 * hi + 8))[0],
                      ((const float4*)(ab + 16 * hi + 8))[1]};
      float4 eq[4] = {((const float4*)(eb + 16 * hi))[0],
                      ((const float4*)(eb + 16 * hi))[1],
                      ((const float4*)(eb + 16 * hi + 8))[0],
                      ((const float4*)(eb + 16 * hi + 8))[1]};
      const float* mv = (const float*)mq;
      const float* ed = (const float*)eq;
      #pragma unroll
      for (int j = 0; j < 16; ++j) {
        float sv = esr + ed[j];
        float lr = (sv > 0.f) ? sv : (ALPHAC * sv);
        ev[s][j] = (mv[j] > 0.f) ? lr : NEGF;
      }
    }

    // Online softmax: combined max over 64 columns.
    float tmax = ev[0][0];
    #pragma unroll
    for (int s = 0; s < 2; ++s)
      #pragma unroll
      for (int j = 0; j < 16; ++j) tmax = fmaxf(tmax, ev[s][j]);
    tmax = fmaxf(tmax, __shfl_xor(tmax, 16, 32));
    float newmax = fmaxf(runmax, tmax);
    float scale  = __expf(runmax - newmax);
    runmax = newmax;

    // Rescale accumulators (D row r+8*hi's scale lives on lane r+8*hi).
    float sc[8];
    #pragma unroll
    for (int r = 0; r < 8; ++r) sc[r] = __shfl(scale, 8 * hi + r, 32);
    #pragma unroll
    for (int t = 0; t < 4; ++t)
      #pragma unroll
      for (int r = 0; r < 8; ++r) acc[t][r] *= sc[r];

    // P @ Wh for both subtiles: hoist B operands, then WMMA chain.
    float psum = 0.f;
    #pragma unroll
    for (int s = 0; s < 2; ++s) {
      v16h P;
      #pragma unroll
      for (int j = 0; j < 16; ++j) {
        float p = __expf(ev[s][j] - newmax);   // masked -> exp(-huge) == 0
        psum += p;
        P[j] = (_Float16)p;
      }
      v16h B[4];
      #pragma unroll
      for (int t = 0; t < 4; ++t) {
        int f = t * 16 + lm;
        v8h b0 = *(const v8h*)&whT[buf][s][f][16 * hi];
        v8h b1 = *(const v8h*)&whT[buf][s][f][16 * hi + 8];
        B[t] = __builtin_shufflevector(b0, b1, 0, 1, 2, 3, 4, 5, 6, 7,
                                       8, 9, 10, 11, 12, 13, 14, 15);
      }
      #pragma unroll
      for (int t = 0; t < 4; ++t) acc[t] = wmma_f16(P, B[t], acc[t]);
    }
    psum += __shfl_xor(psum, 16, 32);
    runsum = runsum * scale + psum;

    __syncthreads();  // everyone done reading buf before it is overwritten
    if (it + 2 < NT) {
      int go = (it + 2) * 8192 + tid * 16;
      uint32_t lb = lds0 + (uint32_t)(buf * 8192);
      async_b128(lb,         go,        gwh);
      async_b128(lb + 4096u, go + 4096, gwh);
    }
  }

  // Normalize, ELU, concat-write out[n][hd*64 + f].
  float dsum[8];
  #pragma unroll
  for (int r = 0; r < 8; ++r) dsum[r] = __shfl(runsum, 8 * hi + r, 32);
  #pragma unroll
  for (int r = 0; r < 8; ++r) {
    int row = m0 + 8 * hi + r;
    #pragma unroll
    for (int t = 0; t < 4; ++t) {
      float v = acc[t][r] / dsum[r];
      v = (v > 0.f) ? v : (__expf(v) - 1.f);
      out[(size_t)row * OUTW + hd * FOUTC + t * 16 + lm] = v;
    }
  }
}

// ---------------------------------------------------------------------------
extern "C" void kernel_launch(void* const* d_in, const int* in_sizes, int n_in,
                              void* d_out, int out_size, void* d_ws, size_t ws_size,
                              hipStream_t stream)
{
  const float* x   = (const float*)d_in[0];
  const float* adj = (const float*)d_in[1];
  const float* Wp[4] = {(const float*)d_in[2], (const float*)d_in[4],
                        (const float*)d_in[6], (const float*)d_in[8]};
  const float* ap[4] = {(const float*)d_in[3], (const float*)d_in[5],
                        (const float*)d_in[7], (const float*)d_in[9]};

  char* ws = (char*)d_ws;
  size_t off = 0;
  auto carve = [&](size_t bytes) -> void* {
    void* p = ws + off;
    off = (off + bytes + 255) & ~(size_t)255;
    return p;
  };
  _Float16* wh = (_Float16*)carve((size_t)NHEADSC * NN * FOUTC * sizeof(_Float16));
  float*    es = (float*)carve((size_t)NHEADSC * NN * sizeof(float));
  float*    ed = (float*)carve((size_t)NHEADSC * NN * sizeof(float));
  float*    h1 = (float*)carve((size_t)NN * OUTW * sizeof(float));
  float*    h2 = (float*)carve((size_t)NN * OUTW * sizeof(float));

  dim3 gw(NN / 64, NHEADSC), bw(128);
  dim3 ga(NN / 128, NHEADSC), ba(256);

  const float* hin = x;
  int fin = 128;
  float* houts[4] = {h1, h2, h1, (float*)d_out};
  for (int l = 0; l < 4; ++l) {
    gat_wh_kernel<<<gw, bw, 0, stream>>>(hin, Wp[l], ap[l], fin, wh, es, ed);
    gat_attn_kernel<<<ga, ba, 0, stream>>>(wh, es, ed, adj, houts[l]);
    hin = houts[l];
    fin = OUTW;
  }
}
